// GIN_82437602279734
// MI455X (gfx1250) — matrix operand order repaired and checked by reading, not compiled, not run on hardware
//
#include <hip/hip_runtime.h>
#include <math.h>

// ---------------------------------------------------------------------------
// GIN forward for MI455X (gfx1250).
//  * All intermediates live in d_ws (~166MB) and fit in the 192MB L2 -> the
//    edge aggregation (the real cost) is L2-atomic bound, not HBM bound.
//  * GEMMs use v_wmma_f32_16x16x32_bf16 (f32 accumulate). One wave owns a
//    16x16 output tile; K is staged through LDS *in fragment order* so each
//    lane's WMMA operand is a single contiguous 32B LDS read (no shuffles).
//  * Kernel is templated on <K, HAS_A2, BN, RELU>: K-loop fully unrolls,
//    the GIN (x+agg) fusion and BatchNorm/ReLU epilogues are branch-free.
// ---------------------------------------------------------------------------

typedef __attribute__((ext_vector_type(16))) __bf16 v16bf;
typedef __attribute__((ext_vector_type(4)))  __bf16 v4bf;
typedef __attribute__((ext_vector_type(8)))  float  v8f;

__device__ __forceinline__ __bf16 to_bf16(float f) {
  unsigned u = __builtin_bit_cast(unsigned, f);
  unsigned r = (u + 0x7FFFu + ((u >> 16) & 1u)) >> 16;   // round-to-nearest-even
  return __builtin_bit_cast(__bf16, (unsigned short)r);
}

#define WPB 8   // waves per block (256 threads = 8 wave32)

// out[row,col] = epilogue( (A[row,:] (+A2[row,:])) @ Wt[:,col] )
// A: [nrows x K] row-stride lda; Wt: [K x M] row-major; out row-stride ldo.
// LDS tiles are stored in WMMA fragment order:
//   A (16x32, ISA 7.12.2): lane = m + 16*((k>>3)&1), slot = (k&7) + 8*((k>>4)&1)
//   B (32x16, same k order as A): lane = n + 16*((k>>3)&1), slot = (k&7) + 8*((k>>4)&1)
template <int K, bool HAS_A2, bool BN, bool RELU>
__global__ __launch_bounds__(256) void wmma_gemm_kernel(
    const float* __restrict__ A,  int lda,
    const float* __restrict__ A2, int lda2,
    const float* __restrict__ Wt, int M,
    const float* __restrict__ bias,
    const float* __restrict__ bng, const float* __restrict__ bnb,
    const float* __restrict__ bnm, const float* __restrict__ bnv,
    float* __restrict__ out, int ldo, int nrows)
{
  __shared__ __align__(32) __bf16 sA[WPB][32][16];
  __shared__ __align__(32) __bf16 sB[WPB][32][16];

  const int lane = threadIdx.x & 31;
  const int w    = threadIdx.x >> 5;
  const int tiles_m = (M + 15) >> 4;
  const int total   = (nrows >> 4) * tiles_m;

  int t = blockIdx.x * WPB + w;
  const bool active = (t < total);
  if (!active) t = total - 1;          // all waves loop (uniform barriers)

  const int row0 = (t / tiles_m) * 16;
  const int col0 = (t % tiles_m) * 16;

  v8f acc = {};

#pragma unroll
  for (int k0 = 0; k0 < K; k0 += 32) {
    __syncthreads();
#pragma unroll
    for (int it = 0; it < 4; ++it) {
      const int idx = lane + 32 * it;
      // ---- stage A: 16 rows x 8 k-quads, float4 loads (global_load_b128)
      {
        const int r = idx >> 3;
        const int c = (idx & 7) * 4;               // k quad base within chunk
        float4 v = *(const float4*)(A + (size_t)(row0 + r) * lda + (k0 + c));
        if (HAS_A2) {
          float4 v2 = *(const float4*)(A2 + (size_t)(row0 + r) * lda2 + (k0 + c));
          v.x += v2.x; v.y += v2.y; v.z += v2.z; v.w += v2.w;
        }
        v4bf p;
        p[0] = to_bf16(v.x); p[1] = to_bf16(v.y);
        p[2] = to_bf16(v.z); p[3] = to_bf16(v.w);
        const int hs   = (c >> 3) & 1;
        const int slot = (c & 7) + ((c >> 4) & 1) * 8;   // quad-contiguous
        *(v4bf*)&sA[w][r + 16 * hs][slot] = p;
      }
      // ---- stage B: 16 cols x 8 k-quads (weights are tiny & L2-hot)
      {
        const int c  = idx >> 3;
        const int kk = (idx & 7) * 4;
        const int col = col0 + c;
        float4 v = make_float4(0.f, 0.f, 0.f, 0.f);
        if (col < M) {
          const float* bp = Wt + (size_t)(k0 + kk) * M + col;
          v.x = bp[0]; v.y = bp[M]; v.z = bp[2 * M]; v.w = bp[3 * M];
        }
        v4bf p;
        p[0] = to_bf16(v.x); p[1] = to_bf16(v.y);
        p[2] = to_bf16(v.z); p[3] = to_bf16(v.w);
        const int hs = (kk >> 3) & 1;
        *(v4bf*)&sB[w][c + 16 * hs][(kk & 7) + ((kk >> 4) & 1) * 8] = p;
      }
    }
    __syncthreads();

    // One contiguous 32B LDS read per operand -> 2x ds_load_b128, no packing.
    v16bf a = *(const v16bf*)&sA[w][lane][0];
    v16bf b = *(const v16bf*)&sB[w][lane][0];
    acc = __builtin_amdgcn_wmma_f32_16x16x32_bf16(
        false, a, false, b, (short)0, acc, false, false);
  }

  // Epilogue. D layout: lane<16 -> row vv, col lane; lane>=16 -> row 8+vv.
  const int hs  = lane >> 4;
  const int col = col0 + (lane & 15);
  if (active && col < M) {
    const float bia = bias ? bias[col] : 0.0f;
    float scale = 1.0f, mean = 0.0f, beta = 0.0f;
    if (BN) {
      scale = bng[col] * rsqrtf(bnv[col] + 1e-5f);
      mean  = bnm[col];
      beta  = bnb[col];
    }
#pragma unroll
    for (int vv = 0; vv < 8; ++vv) {
      float h = acc[vv] + bia;
      if (BN)   h = (h - mean) * scale + beta;
      if (RELU) h = fmaxf(h, 0.0f);
      out[(size_t)(row0 + vv + 8 * hs) * ldo + col] = h;
    }
  }
}

// Edge scatter-add: one thread per (edge, float4 chunk). Coalesced float4
// gather from the source row, 4x global_atomic_add_f32 into the destination
// row. All traffic is L2-resident (working set < 192MB).
__global__ __launch_bounds__(256) void scatter_add_kernel(
    const float* __restrict__ x, int ldx,
    const int* __restrict__ ei, int E, int d4,
    float* __restrict__ agg, int lda)
{
  long long idx = (long long)blockIdx.x * blockDim.x + threadIdx.x;
  long long total = (long long)E * d4;
  if (idx >= total) return;
  int e = (int)(idx / d4);
  int c = (int)(idx % d4) * 4;
  int s = ei[e];
  int t = ei[E + e];
  const float4 v = *(const float4*)(x + (size_t)s * ldx + c);
  float* d = agg + (size_t)t * lda + c;
  atomicAdd(d + 0, v.x);
  atomicAdd(d + 1, v.y);
  atomicAdd(d + 2, v.z);
  atomicAdd(d + 3, v.w);
}

__global__ __launch_bounds__(256) void logsoftmax_kernel(
    const float* __restrict__ h, float* __restrict__ out, int n, int d)
{
  int i = blockIdx.x * blockDim.x + threadIdx.x;
  if (i >= n) return;
  const float* r = h + (size_t)i * d;
  float mx = -INFINITY;
  for (int j = 0; j < d; ++j) mx = fmaxf(mx, r[j]);
  float s = 0.0f;
  for (int j = 0; j < d; ++j) s += expf(r[j] - mx);
  float l = logf(s) + mx;
  float* o = out + (size_t)i * d;
  for (int j = 0; j < d; ++j) o[j] = r[j] - l;
}

template <int K, bool HAS_A2, bool BN, bool RELU>
static inline void launch_gemm(const float* A, int lda, const float* A2, int lda2,
                               const float* Wt, int M, const float* bias,
                               const float* g, const float* be, const float* m,
                               const float* v, float* out, int ldo, int n,
                               hipStream_t s)
{
  int tiles  = (n / 16) * ((M + 15) / 16);
  int blocks = (tiles + WPB - 1) / WPB;
  hipLaunchKernelGGL((wmma_gemm_kernel<K, HAS_A2, BN, RELU>),
                     dim3(blocks), dim3(256), 0, s,
                     A, lda, A2, lda2, Wt, M, bias, g, be, m, v, out, ldo, n);
}

static inline void launch_scatter(const float* x, int ldx, const int* ei, int E,
                                  int d, float* agg, int lda, hipStream_t s)
{
  int d4 = d / 4;
  long long total = (long long)E * d4;
  int blocks = (int)((total + 255) / 256);
  hipLaunchKernelGGL(scatter_add_kernel, dim3(blocks), dim3(256), 0, s,
                     x, ldx, ei, E, d4, agg, lda);
}

extern "C" void kernel_launch(void* const* d_in, const int* in_sizes, int n_in,
                              void* d_out, int out_size, void* d_ws, size_t ws_size,
                              hipStream_t stream) {
  const float* x      = (const float*)d_in[0];
  const int*   ei     = (const int*)  d_in[1];
  const float* W      = (const float*)d_in[2];
  const float* c1_W1  = (const float*)d_in[3];
  const float* c1_b1  = (const float*)d_in[4];
  const float* c1_g   = (const float*)d_in[5];
  const float* c1_be  = (const float*)d_in[6];
  const float* c1_m   = (const float*)d_in[7];
  const float* c1_v   = (const float*)d_in[8];
  const float* c1_W2  = (const float*)d_in[9];
  const float* c1_b2  = (const float*)d_in[10];
  const float* cW1    = (const float*)d_in[11];
  const float* cb1    = (const float*)d_in[12];
  const float* cg     = (const float*)d_in[13];
  const float* cbe    = (const float*)d_in[14];
  const float* cm     = (const float*)d_in[15];
  const float* cv     = (const float*)d_in[16];
  const float* cW2    = (const float*)d_in[17];
  const float* cb2    = (const float*)d_in[18];
  const float* lin1_W = (const float*)d_in[19];
  const float* lin1_b = (const float*)d_in[20];
  const float* lin2_W = (const float*)d_in[21];
  const float* lin2_b = (const float*)d_in[22];

  const int N = in_sizes[0] / 128;   // 100000 (divisible by 16)
  const int E = in_sizes[1] / 2;     // 3200000

  // Workspace layout (floats); ~166MB, L2-resident on MI455X.
  float* xp    = (float*)d_ws;                 // [N x 128] input projection
  float* agg   = xp   + (size_t)N * 128;       // [N x 128] aggregation scratch
  float* hcat  = agg  + (size_t)N * 128;       // [N x 128] = [h1|h2|h3|h4]
  float* hmid  = hcat + (size_t)N * 128;       // [N x 32]  MLP hidden
  float* lin1o = xp;                           // reuse xp after conv1

  float* ho  = (float*)d_out;                  // [N x 40] logits
  float* lsm = ho + (size_t)N * 40;            // [N x 40] log_softmax

  // xp = x @ W
  launch_gemm<128, false, false, false>(x, 128, nullptr, 0, W, 128,
      nullptr, nullptr, nullptr, nullptr, nullptr, xp, 128, N, stream);

  // conv1 (d=128 -> 32 -> 32), h1 -> hcat[:, 0:32]
  hipMemsetAsync(agg, 0, (size_t)N * 128 * sizeof(float), stream);
  launch_scatter(xp, 128, ei, E, 128, agg, 128, stream);
  launch_gemm<128, true, true, true>(xp, 128, agg, 128, c1_W1, 32,
      c1_b1, c1_g, c1_be, c1_m, c1_v, hmid, 32, N, stream);
  launch_gemm<32, false, false, true>(hmid, 32, nullptr, 0, c1_W2, 32,
      c1_b2, nullptr, nullptr, nullptr, nullptr, hcat, 128, N, stream);

  // conv2..conv4 (32 -> 32 -> 32), h_{i+2} -> hcat[:, 32(i+1) : 32(i+2)]
  for (int i = 0; i < 3; ++i) {
    const float* hin = hcat + (size_t)i * 32;          // col offset, stride 128
    hipMemsetAsync(agg, 0, (size_t)N * 32 * sizeof(float), stream);
    launch_scatter(hin, 128, ei, E, 32, agg, 32, stream);
    launch_gemm<32, true, true, true>(hin, 128, agg, 32,
        cW1 + (size_t)i * 32 * 32, 32,
        cb1 + i * 32, cg + i * 32, cbe + i * 32, cm + i * 32, cv + i * 32,
        hmid, 32, N, stream);
    launch_gemm<32, false, false, true>(hmid, 32, nullptr, 0,
        cW2 + (size_t)i * 32 * 32, 32,
        cb2 + i * 32, nullptr, nullptr, nullptr, nullptr,
        hcat + (size_t)(i + 1) * 32, 128, N, stream);
  }

  // lin1: relu(hcat @ lin1_W + b)  [N x 128]
  launch_gemm<128, false, false, true>(hcat, 128, nullptr, 0, lin1_W, 128,
      lin1_b, nullptr, nullptr, nullptr, nullptr, lin1o, 128, N, stream);

  // lin2: logits [N x 40] -> d_out (no relu)
  launch_gemm<128, false, false, false>(lin1o, 128, nullptr, 0, lin2_W, 40,
      lin2_b, nullptr, nullptr, nullptr, nullptr, ho, 40, N, stream);

  // log_softmax -> second half of d_out
  hipLaunchKernelGGL(logsoftmax_kernel, dim3((N + 255) / 256), dim3(256), 0, stream,
                     ho, lsm, N, 40);
}
